// TreeModel_25220047962304
// MI455X (gfx1250) — matrix operand order repaired and checked by reading, not compiled, not run on hardware
//
#include <hip/hip_runtime.h>

typedef __attribute__((ext_vector_type(16))) _Float16 v16h;
typedef __attribute__((ext_vector_type(2)))  _Float16 v2h;
typedef __attribute__((ext_vector_type(8)))  float    v8f;
typedef __attribute__((ext_vector_type(4)))  int      v4i;
typedef __attribute__((ext_vector_type(2)))  int      v2i;

typedef __attribute__((address_space(1))) v4i gv4i;   // global int4
typedef __attribute__((address_space(3))) v4i lv4i;   // LDS int4

namespace {
constexpr int kBRows  = 16384;
constexpr int kF      = 1024;
constexpr int kNInner = 511;
constexpr int kNCols  = 512;
constexpr int kNLeaf  = 512;
constexpr int kNCls   = 10;

constexpr int kMT      = 128;   // rows per block (16 waves)
constexpr int kKT      = 32;    // K per stage step (one f16 WMMA K)
constexpr int kStrideD = 20;    // dwords per staged row: 16 data + 4 pad
                                // (20*k mod 64 hits all banks; 16B aligned -> b128)
constexpr int kPStride = 520;   // p-tile stride (floats)

constexpr int kLeafBytes = kNLeaf * kNCls * 4;     // 20480
constexpr int kAsBytes   = kMT   * kStrideD * 4;   // 10240
constexpr int kBsBytes   = kNCols* kStrideD * 4;   // 40960
constexpr int kAs0 = kLeafBytes;
constexpr int kAs1 = kAs0 + kAsBytes;
constexpr int kBs0 = kAs1 + kAsBytes;
constexpr int kBs1 = kBs0 + kBsBytes;              // staging ends @ 122880
constexpr int kPBytes    = kMT * kPStride * 4;     // 266240 (union over staging)
constexpr int kSmemBytes = kLeafBytes + kPBytes;   // 286720 < 320KB
constexpr int kIters = kF / kKT;                   // 32
} // namespace

__device__ __forceinline__ int packh2(float a, float b) {
  v2h t; t[0] = (_Float16)a; t[1] = (_Float16)b;
  return __builtin_bit_cast(int, t);
}

// async global->LDS b128 copy (ASYNCcnt); g and ldsByteAddr must be 16B aligned
__device__ __forceinline__ void async_b128(const void* g, unsigned ldsByteAddr) {
#if __has_builtin(__builtin_amdgcn_global_load_async_to_lds_b128)
  __builtin_amdgcn_global_load_async_to_lds_b128(
      (gv4i*)(unsigned long long)g, (lv4i*)ldsByteAddr, 0, 0);
#else
  const unsigned long long ga = (unsigned long long)g;
  asm volatile("global_load_async_to_lds_b128 %0, %1, off"
               :: "v"(ldsByteAddr), "v"(ga) : "memory");
#endif
}

__device__ __forceinline__ void wait_async0() {
#if __has_builtin(__builtin_amdgcn_s_wait_asynccnt)
  __builtin_amdgcn_s_wait_asynccnt(0);
#else
  asm volatile("s_wait_asynccnt 0x0" ::: "memory");
#endif
}

// ---- kernel 0: W (1024x511 f32, k-major) -> Wh (512x1024 f16, n-major), col 511 zeroed
__global__ __launch_bounds__(256)
void convert_W(const float* __restrict__ W, _Float16* __restrict__ Wh) {
  __shared__ float tile[64][65];
  const int k0 = (blockIdx.x & 15) * 64;
  const int n0 = (blockIdx.x >> 4) * 64;
  const int t  = threadIdx.x;
  {
    const int j = t & 63, i0 = t >> 6;          // j = n (coalesced), i = k
    #pragma unroll
    for (int s = 0; s < 16; ++s) {
      const int i = i0 + 4 * s;
      const int n = n0 + j;
      tile[i][j] = (n < kNInner) ? W[(size_t)(k0 + i) * kNInner + n] : 0.0f;
    }
  }
  __syncthreads();
  {
    const int i = t & 63, j0 = t >> 6;          // i = k (coalesced out), j = n
    #pragma unroll
    for (int s = 0; s < 16; ++s) {
      const int j = j0 + 4 * s;
      Wh[(size_t)(n0 + j) * kF + (k0 + i)] = (_Float16)tile[i][j];
    }
  }
}

// ---- fused: p = sigmoid(x@W+b) via WMMA, tree path products, @leaf_logits
__global__ __launch_bounds__(512, 1)
void soft_tree_fused(const float* __restrict__ x,
                     const _Float16* __restrict__ Wh,
                     const float* __restrict__ bias,
                     const float* __restrict__ leaf,
                     float* __restrict__ out)
{
  extern __shared__ char smem[];
  float* leafLds = reinterpret_cast<float*>(smem);
  float* pLds    = reinterpret_cast<float*>(smem + kLeafBytes);

  const int tid   = threadIdx.x;
  const int wid   = tid >> 5;
  const int lane  = tid & 31;
  const int nl    = lane & 15;
  const int hi    = lane >> 4;
  const int waveM = wid >> 3;          // 0..1 -> 64-row strip (4 M-tiles)
  const int waveN = wid & 7;           // 0..7 -> 64-col strip (4 N-tiles)
  const int rowBase = blockIdx.x * kMT;

  for (int i = tid; i < kNLeaf * kNCls; i += 512) leafLds[i] = leaf[i];

  v8f acc[16];                          // acc[mt*4+nt]
  #pragma unroll
  for (int t = 0; t < 16; ++t)
    #pragma unroll
    for (int r = 0; r < 8; ++r) acc[t][r] = 0.0f;

  // A staging: global f32 -> regs -> (cvt f16) -> LDS; B staging: async global->LDS
  const int ar  = tid >> 2;            // A row 0..127
  const int ac4 = tid & 3;             // A col group
  const int bn  = tid;                 // B n-row 0..511

  const char* bsrcBase = (const char*)Wh + (size_t)bn * (kF * 2);
  const unsigned ldsB0 = (unsigned)(unsigned long long)(smem + kBs0) + bn * (kStrideD * 4);
  const unsigned ldsB1 = (unsigned)(unsigned long long)(smem + kBs1) + bn * (kStrideD * 4);

  float4 areg[2];

  auto gloadA = [&](int it) {
    const int kb = it * kKT;
    const float* asrc = x + (size_t)(rowBase + ar) * kF + kb;
    areg[0] = *reinterpret_cast<const float4*>(asrc + ac4 * 4);
    areg[1] = *reinterpret_cast<const float4*>(asrc + (ac4 + 4) * 4);
  };
  auto sstoreA = [&](int buf) {
    unsigned* AsU = reinterpret_cast<unsigned*>(smem + (buf ? kAs1 : kAs0));
    v2i a0; a0[0] = packh2(areg[0].x, areg[0].y); a0[1] = packh2(areg[0].z, areg[0].w);
    v2i a1; a1[0] = packh2(areg[1].x, areg[1].y); a1[1] = packh2(areg[1].z, areg[1].w);
    *reinterpret_cast<v2i*>(AsU + ar * kStrideD + 2 * ac4)     = a0;
    *reinterpret_cast<v2i*>(AsU + ar * kStrideD + 8 + 2 * ac4) = a1;
  };
  auto asyncB = [&](int it, int buf) {
    const char* g = bsrcBase + it * (kKT * 2);     // 64B per K-step
    const unsigned l = buf ? ldsB1 : ldsB0;
    #pragma unroll
    for (int j = 0; j < 4; ++j) async_b128(g + 16 * j, l + 16 * j);
  };

  gloadA(0);
  asyncB(0, 0);
  sstoreA(0);

  for (int it = 0; it < kIters; ++it) {
    if (it + 1 < kIters) gloadA(it + 1);     // global loads: LOADcnt
    wait_async0();                            // buf[it&1] B-tile landed
    __syncthreads();                          // + all reads of buf[(it+1)&1] done
    if (it + 1 < kIters) asyncB(it + 1, (it + 1) & 1);  // overlap with wmmas

    const unsigned* AsU = reinterpret_cast<const unsigned*>(smem + ((it & 1) ? kAs1 : kAs0));
    const unsigned* BsU = reinterpret_cast<const unsigned*>(smem + ((it & 1) ? kBs1 : kBs0));

    // 4 A fragments (one per M-tile): dwords row*20 + {0..3,8..11} + 4*hi
    v16h afr[4];
    #pragma unroll
    for (int mt = 0; mt < 4; ++mt) {
      const v4i* ap = reinterpret_cast<const v4i*>(
          AsU + (waveM * 64 + mt * 16 + nl) * kStrideD + 4 * hi);
      afr[mt] = __builtin_bit_cast(
          v16h, __builtin_shufflevector(ap[0], ap[2], 0, 1, 2, 3, 4, 5, 6, 7));
    }
    // 4 B fragments (one per N-tile): dwords col*20 + {0..3,8..11} + 4*hi
    v16h bfr[4];
    #pragma unroll
    for (int nt = 0; nt < 4; ++nt) {
      const v4i* bp = reinterpret_cast<const v4i*>(
          BsU + (waveN * 64 + nt * 16 + nl) * kStrideD + 4 * hi);
      bfr[nt] = __builtin_bit_cast(
          v16h, __builtin_shufflevector(bp[0], bp[2], 0, 1, 2, 3, 4, 5, 6, 7));
    }
    // 16 wmmas with full A/B fragment reuse: 16B LDS per wmma per lane
    #pragma unroll
    for (int mt = 0; mt < 4; ++mt)
      #pragma unroll
      for (int nt = 0; nt < 4; ++nt)
        acc[mt * 4 + nt] = __builtin_amdgcn_wmma_f32_16x16x32_f16(
            false, afr[mt], false, bfr[nt], (short)0, acc[mt * 4 + nt], false, false);

    if (it + 1 < kIters) sstoreA((it + 1) & 1);  // A ds stores AFTER wmmas
  }

  __syncthreads();   // staging reads done; LDS union becomes the p tile

  // ---------------- bias + sigmoid -> p tile in LDS ----------------
  #pragma unroll
  for (int mt = 0; mt < 4; ++mt) {
    #pragma unroll
    for (int nt = 0; nt < 4; ++nt) {
      const int col = waveN * 64 + nt * 16 + nl;
      const float bb = (col < kNInner) ? bias[col] : 0.0f;
      #pragma unroll
      for (int r = 0; r < 8; ++r) {
        const float t  = acc[mt * 4 + nt][r] + bb;
        const float pv = 1.0f / (1.0f + __expf(-t));
        const int   m  = waveM * 64 + mt * 16 + hi * 8 + r;  // C/D: M = vgpr + 8*hi
        pLds[m * kPStride + col] = pv;
      }
    }
  }

  __syncthreads();

  // ---------------- tree path products + leaf-logit GEMM ----------------
  {
    const int row   = tid >> 2;    // 0..127
    const int chunk = tid & 3;
    const float* prow = pLds + row * kPStride;

    float acc2[kNCls];
    #pragma unroll
    for (int c = 0; c < kNCls; ++c) acc2[c] = 0.0f;

    for (int li = 0; li < 128; ++li) {
      const int leafI = li * 4 + chunk;   // interleave: node reads broadcast
      float pr = 1.0f;
      #pragma unroll
      for (int l = 0; l <= 8; ++l) {
        const int node = (1 << l) - 1 + (leafI >> (9 - l));  // heap index <= 510
        const float pv = prow[node];
        const int bit  = (leafI >> (8 - l)) & 1;
        pr *= bit ? pv : (1.0f - pv);
      }
      const float* ll = leafLds + leafI * kNCls;
      #pragma unroll
      for (int c = 0; c < kNCls; ++c) acc2[c] += pr * ll[c];
    }

    #pragma unroll
    for (int c = 0; c < kNCls; ++c) {
      acc2[c] += __shfl_xor(acc2[c], 1, 32);
      acc2[c] += __shfl_xor(acc2[c], 2, 32);
    }
    if (chunk == 0) {
      float* orow = out + (size_t)(rowBase + row) * kNCls;
      #pragma unroll
      for (int c = 0; c < kNCls; ++c) orow[c] = acc2[c];
    }
  }
}

extern "C" void kernel_launch(void* const* d_in, const int* in_sizes, int n_in,
                              void* d_out, int out_size, void* d_ws, size_t ws_size,
                              hipStream_t stream) {
  (void)in_sizes; (void)n_in; (void)out_size; (void)ws_size;
  const float* x    = (const float*)d_in[0];
  const float* W    = (const float*)d_in[1];
  const float* bias = (const float*)d_in[2];
  const float* leaf = (const float*)d_in[3];
  float* out = (float*)d_out;
  _Float16* Wh = (_Float16*)d_ws;           // 512*1024*2 = 1 MB scratch

  convert_W<<<dim3(128), dim3(256), 0, stream>>>(W, Wh);
  soft_tree_fused<<<dim3(kBRows / kMT), dim3(512), kSmemBytes, stream>>>(
      x, Wh, bias, leaf, out);
}
